// NeighborGatedAttention_45062796870083
// MI455X (gfx1250) — matrix operand order repaired and checked by reading, not compiled, not run on hardware
//
#include <hip/hip_runtime.h>

// ---------------------------------------------------------------------------
// NeighborGatedAttention fused kernel for gfx1250 (MI455X), wave32 + WMMA.
//  - one workgroup (256 thr / 8 waves) per batch row b
//  - all 4 matmuls per layer via v_wmma_f32_16x16x32_f16, f32 accumulate
//  - weights pre-converted+transposed to f16 in d_ws by a prep kernel, then
//    DMA'd into LDS per-block via the Tensor Data Mover (tensor_load_to_lds,
//    s_wait_tensorcnt), overlapped with the surrounding compute stages
// ---------------------------------------------------------------------------

typedef __attribute__((ext_vector_type(16))) _Float16 v16h;
typedef __attribute__((ext_vector_type(8)))  _Float16 v8h;
typedef __attribute__((ext_vector_type(4)))  _Float16 v4h;
typedef __attribute__((ext_vector_type(8)))  float    v8f;
typedef __attribute__((ext_vector_type(4)))  float    v4f;
typedef __attribute__((ext_vector_type(4)))  unsigned int u32x4;
typedef __attribute__((ext_vector_type(8)))  int      i32x8;
typedef __attribute__((ext_vector_type(4)))  int      i32x4;

constexpr int LAYERS  = 2;
constexpr int NNEI    = 120;
constexpr int EE      = 128;   // embedding
constexpr int HH      = 128;   // hidden (NH=1, HD=128)
constexpr int NP      = 128;   // padded rows (120 -> 128)
constexpr int K3H     = 384;   // 3*H
constexpr int THREADS = 256;
constexpr int NB      = 4096;  // batch rows

constexpr float SCALING     = 0.08838834764831845f; // 128^-0.5
constexpr float ATTNW_SHIFT = 20.0f;
constexpr float LN_EPS      = 1e-5f;

// LDS layout (dynamic):
//   [     0 ..  32768) R0: xh (f16 128x128)   -> later s (f16 scores)
//   [ 32768 ..  65536) R1: qh (f16 128x128)   -> later ah (f16 attn)
//   [ 65536 ..  98304) R2: kh (f16 128x128)   -> later oh (f16 o)
//   [ 98304 .. 131072) R3: vT (f16 128x128, d-major)
//   ybuf (f32 128x128 = 64KB) overlays R0+R1 during out-proj / LN
//   [131072 .. 133632) rx/ry/rz/swl/maskf (5 x 128 f32)
//   [133632 .. 231936) WtL: W_in^T f16 [384][128] (TDM dest); reused for
//                      W_out^T f16 [128][128] after stage 2
constexpr int    R0_OFF = 0;
constexpr int    R1_OFF = 32768;
constexpr int    R2_OFF = 65536;
constexpr int    R3_OFF = 98304;
constexpr int    SM_OFF = 131072;
constexpr int    WT_OFF = SM_OFF + 5 * NP * (int)sizeof(float);   // 133632
constexpr size_t SMEM_BYTES = (size_t)WT_OFF + (size_t)K3H * EE * 2; // 231936

#if defined(__AMDGCN__) && __has_builtin(__builtin_amdgcn_tensor_load_to_lds)
#define HAVE_TDM 1
#else
#define HAVE_TDM 0
#endif

// --- WMMA fragment loaders (layouts per CDNA5 ISA 7.12.2) -------------------
// A (16x32 f16): lanes 0-15 = rows M0..15 holding K {0..7,16..23};
//                lanes 16-31 = rows M0..15 holding K {8..15,24..31}.
__device__ __forceinline__ v16h frag_a(const _Float16* base, int row0, int k0,
                                       int stride, int lane) {
  const _Float16* p = base + (row0 + (lane & 15)) * stride + k0 + ((lane >> 4) << 3);
  v8h lo = *(const v8h*)(p);
  v8h hi = *(const v8h*)(p + 16);
  return __builtin_shufflevector(lo, hi, 0,1,2,3,4,5,6,7,8,9,10,11,12,13,14,15);
}
// B (32x16 f16), loaded from B^T stored row-major (row = column-of-B, len K):
// lanes 0-15 = cols N0..15 holding K {0..7,16..23}; lanes 16-31 hold K {8..15,24..31}.
__device__ __forceinline__ v16h frag_b(const _Float16* base, int col0, int k0,
                                       int stride, int lane) {
  const _Float16* p = base + (col0 + (lane & 15)) * stride + k0 + ((lane >> 4) << 3);
  v8h lo = *(const v8h*)(p);
  v8h hi = *(const v8h*)(p + 16);
  return __builtin_shufflevector(lo, hi, 0,1,2,3,4,5,6,7,8,9,10,11,12,13,14,15);
}

// K=128 accumulate with pre-hoisted A fragments (4 x K32 steps).
__device__ __forceinline__ v8f wmma_k128(const v16h a[4], const _Float16* Bt,
                                         int n0, int strideB, int lane) {
  v8f c = {};
#pragma unroll
  for (int kk = 0; kk < 4; ++kk) {
    v16h bf = frag_b(Bt, n0, kk * 32, strideB, lane);
    c = __builtin_amdgcn_wmma_f32_16x16x32_f16(false, a[kk], false, bf,
                                               (short)0, c, false, false);
  }
  return c;
}

// --- Tensor Data Mover: DMA a [rows x 128] f16 row-major block into LDS -----
// D# built per CDNA5 ISA sec.8 (count=1, type=2,
// data_size=2B, 2D tensor dim0=128 stride=128, tile = rows x 128).
// clang-23 toolchain: 6-arg builtin (g0 x4u, g1 x8i, g2 x4i, g3 x4i, x8i, cpol)
__device__ __forceinline__ void weights_issue(const _Float16* gsrc, _Float16* ldst,
                                              int rows, int tid, int wave) {
#if HAVE_TDM
  if (wave == 0) {
    unsigned long long ga = (unsigned long long)(uintptr_t)gsrc;
    unsigned lds = (unsigned)(uintptr_t)ldst;  // flat LDS aperture: low 32b = offset
    u32x4 g0;
    g0[0] = 1u;                                   // count=1, user mode
    g0[1] = lds;                                  // lds_addr
    g0[2] = (unsigned)ga;                         // global_addr[31:0]
    g0[3] = (unsigned)((ga >> 32) & 0x1FFFFFFu) | (2u << 30); // addr[56:32]|type=2
    i32x8 g1;
    g1[0] = (int)(1u << 16);                      // data_size=1 -> 2 bytes
    g1[1] = (int)((unsigned)EE << 16);            // tensor_dim0[15:0] @bit48
    g1[2] = (int)((unsigned)rows << 16);          // dim0 hi=0 | tensor_dim1 lo
    g1[3] = (int)((unsigned)EE << 16);            // dim1 hi=0 | tile_dim0=128
    g1[4] = (int)(rows & 0xFFFF);                 // tile_dim1 | tile_dim2=0
    g1[5] = (int)EE;                              // tensor_dim0_stride = 128
    g1[6] = 0;
    g1[7] = 0;
    i32x4 z4 = {};
    i32x8 z8 = {};
    __builtin_amdgcn_tensor_load_to_lds(g0, g1, z4, z4, z8, 0);
  }
#else
  const int n = rows * EE;
  for (int idx = tid * 8; idx < n; idx += THREADS * 8)
    *(v8h*)(ldst + idx) = *(const v8h*)(gsrc + idx);
#endif
}
__device__ __forceinline__ void weights_wait(int wave) {
#if HAVE_TDM
  if (wave == 0) __builtin_amdgcn_s_wait_tensorcnt(0);
#endif
}

// --- weight prep: f32 -> f16, transposed so B^T rows are contiguous ---------
__global__ __launch_bounds__(256) void nga_prep(const float* __restrict__ W_in,
                                                const float* __restrict__ W_out,
                                                _Float16* __restrict__ wtin,
                                                _Float16* __restrict__ wot) {
  int idx = blockIdx.x * blockDim.x + threadIdx.x;
  const int total1 = LAYERS * K3H * EE;  // wtin[l][n][e] = W_in[l][e][n]
  const int total2 = LAYERS * EE * HH;   // wot [l][e][h] = W_out[l][h][e]
  if (idx < total1) {
    int l = idx / (K3H * EE), r = idx % (K3H * EE);
    int n = r / EE, e = r % EE;
    wtin[idx] = (_Float16)W_in[(l * EE + e) * K3H + n];
  } else if (idx < total1 + total2) {
    int i2 = idx - total1;
    int l = i2 / (EE * HH), r = i2 % (EE * HH);
    int e = r / HH, h = r % HH;
    wot[i2] = (_Float16)W_out[(l * HH + h) * EE + e];
  }
}

// --- fused attention kernel: one block per batch row ------------------------
__global__ __launch_bounds__(THREADS) void nga_main(
    const float* __restrict__ input_G, const int* __restrict__ nei_mask,
    const float* __restrict__ input_r, const float* __restrict__ sw,
    const float* __restrict__ b_in, const float* __restrict__ b_out,
    const float* __restrict__ ln_g, const float* __restrict__ ln_b,
    const _Float16* __restrict__ wtin, const _Float16* __restrict__ wot,
    float* __restrict__ outp) {
  extern __shared__ __align__(16) char smem[];
  _Float16* R0 = (_Float16*)(smem + R0_OFF);   // xh / s
  _Float16* R1 = (_Float16*)(smem + R1_OFF);   // qh / ah
  _Float16* R2 = (_Float16*)(smem + R2_OFF);   // kh / oh
  _Float16* R3 = (_Float16*)(smem + R3_OFF);   // vT
  float*  ybuf = (float*)(smem);               // 64KB over R0+R1
  float*  rx   = (float*)(smem + SM_OFF);
  float*  ry   = rx + NP;
  float*  rz   = ry + NP;
  float*  swl  = rz + NP;
  float*  mkf  = swl + NP;
  _Float16* WL = (_Float16*)(smem + WT_OFF);   // WtL (384x128) / WoL (128x128)

  const int b    = blockIdx.x;
  const int tid  = threadIdx.x;
  const int lane = tid & 31;
  const int wave = tid >> 5;

  // per-row small tensors (constant across layers)
  for (int i = tid; i < NP; i += THREADS) {
    if (i < NNEI) {
      const float* rp = input_r + (size_t)(b * NNEI + i) * 3;
      rx[i] = rp[0]; ry[i] = rp[1]; rz[i] = rp[2];
      swl[i] = sw[b * NNEI + i];
      mkf[i] = (nei_mask[b * NNEI + i] != 0) ? 1.0f : 0.0f;
    } else {
      rx[i] = ry[i] = rz[i] = 0.0f; swl[i] = 0.0f; mkf[i] = 0.0f;
    }
  }

  for (int l = 0; l < LAYERS; ++l) {
    const float*    xsrc = (l == 0) ? input_G : outp;   // block-local rows only
    const float*    bi   = b_in  + l * K3H;
    const float*    bo   = b_out + l * EE;
    const float*    lng  = ln_g  + l * EE;
    const float*    lnb  = ln_b  + l * EE;
    const _Float16* Wt   = wtin + (size_t)l * K3H * EE; // [384][128]
    const _Float16* Wo   = wot  + (size_t)l * EE * HH;  // [128][128]
    const float*    xrow = xsrc + (size_t)b * NNEI * EE;

    __syncthreads();  // orders prior-layer global writes & LDS reuse

    // TDM: start streaming W_in^T into LDS; overlaps with stage 1
    weights_issue(Wt, WL, K3H, tid, wave);

    // ---- stage 1: x -> LDS f16 (pad rows 120..127 with 0), 16B vectors ----
    for (int idx4 = tid; idx4 < NP * EE / 4; idx4 += THREADS) {
      const int i = idx4 >> 5, c = idx4 & 31;
      v4h o = {};
      if (i < NNEI) {
        v4f v = *(const v4f*)(xrow + i * EE + c * 4);
#pragma unroll
        for (int u = 0; u < 4; ++u) o[u] = (_Float16)v[u];
      }
      *(v4h*)(R0 + i * EE + c * 4) = o;
    }
    __syncthreads();
    weights_wait(wave);        // wave 0: s_wait_tensorcnt 0
    __syncthreads();

    // ---- stage 2: qkv = x @ W_in + b_in ; wave owns M-tile strip mt=wave ----
    {
      const int mt = wave;
      v16h a[4];
#pragma unroll
      for (int kk = 0; kk < 4; ++kk) a[kk] = frag_a(R0, mt * 16, kk * 32, EE, lane);
      for (int nt = 0; nt < K3H / 16; ++nt) {
        v8f c = wmma_k128(a, WL, nt * 16, EE, lane);
        const int n  = nt * 16 + (lane & 15);
        const int mb = mt * 16 + ((lane >> 4) << 3);
        const float bn = bi[n];
#pragma unroll
        for (int vg = 0; vg < 8; ++vg) {
          const int m = mb + vg;
          const float val = c[vg] + bn;
          if (n < HH)            R1[m * EE + n]            = (_Float16)val;  // q
          else if (n < 2 * HH)   R2[m * EE + (n - HH)]     = (_Float16)val;  // k
          else                   R3[(n - 2 * HH) * NP + m] = (_Float16)val;  // v^T
        }
      }
    }
    __syncthreads();

    // TDM: start streaming W_out^T into LDS (reuses WL); overlaps stages 3-6
    weights_issue(Wo, WL, EE, tid, wave);

    // ---- stage 3: l2-normalize. thr 0-127: q,k rows; thr 128-255: v cols ----
    if (tid < NP) {
      const int i = tid;
      v8h* qr = (v8h*)(R1 + i * EE);
      float ss = 0.f;
#pragma unroll 4
      for (int c = 0; c < 16; ++c) {
        v8h v = qr[c];
#pragma unroll
        for (int u = 0; u < 8; ++u) { float f = (float)v[u]; ss = fmaf(f, f, ss); }
      }
      float sc = SCALING / fmaxf(sqrtf(ss), 1e-12f);
#pragma unroll 4
      for (int c = 0; c < 16; ++c) {
        v8h v = qr[c];
#pragma unroll
        for (int u = 0; u < 8; ++u) v[u] = (_Float16)((float)v[u] * sc);
        qr[c] = v;
      }
      v8h* kr = (v8h*)(R2 + i * EE);
      ss = 0.f;
#pragma unroll 4
      for (int c = 0; c < 16; ++c) {
        v8h v = kr[c];
#pragma unroll
        for (int u = 0; u < 8; ++u) { float f = (float)v[u]; ss = fmaf(f, f, ss); }
      }
      sc = 1.0f / fmaxf(sqrtf(ss), 1e-12f);
#pragma unroll 4
      for (int c = 0; c < 16; ++c) {
        v8h v = kr[c];
#pragma unroll
        for (int u = 0; u < 8; ++u) v[u] = (_Float16)((float)v[u] * sc);
        kr[c] = v;
      }
    } else {
      const int j = tid - NP;  // v column j of vT[d][j]
      float ss = 0.f;
      for (int d = 0; d < HH; ++d) { float f = (float)R3[d * NP + j]; ss = fmaf(f, f, ss); }
      const float sc = 1.0f / fmaxf(sqrtf(ss), 1e-12f);
      for (int d = 0; d < HH; ++d) R3[d * NP + j] = (_Float16)((float)R3[d * NP + j] * sc);
    }
    __syncthreads();

    // ---- stage 4: scores s = q @ k^T  (B^T = k row-major) -> R0 (f16) ----
    {
      const int mt = wave;
      v16h a[4];
#pragma unroll
      for (int kk = 0; kk < 4; ++kk) a[kk] = frag_a(R1, mt * 16, kk * 32, EE, lane);
      for (int nt = 0; nt < NP / 16; ++nt) {
        v8f c = wmma_k128(a, R2, nt * 16, EE, lane);
        const int n  = nt * 16 + (lane & 15);
        const int mb = mt * 16 + ((lane >> 4) << 3);
#pragma unroll
        for (int vg = 0; vg < 8; ++vg) R0[(mb + vg) * NP + n] = (_Float16)c[vg];
      }
    }
    __syncthreads();

    // ---- stage 5: gated softmax per row i -> ah in R1 (f16, padded 0) ----
    if (tid < NP) {
      const int i = tid;
      const v8h* srow = (const v8h*)(R0 + i * NP);
      v8h*       arow = (v8h*)(R1 + i * NP);
      const v8h  zz = {};
      if (i < NNEI) {
        const float swi = swl[i];
        float mx = -1e30f;
        for (int c = 0; c < NNEI / 8; ++c) {
          v8h sv = srow[c];
#pragma unroll
          for (int u = 0; u < 8; ++u) {
            float t = ((float)sv[u] + ATTNW_SHIFT) * swi * swl[c * 8 + u] - ATTNW_SHIFT;
            mx = fmaxf(mx, t);
          }
        }
        float den = 0.f;
        for (int c = 0; c < NNEI / 8; ++c) {
          v8h sv = srow[c];
#pragma unroll
          for (int u = 0; u < 8; ++u) {
            float t = ((float)sv[u] + ATTNW_SHIFT) * swi * swl[c * 8 + u] - ATTNW_SHIFT;
            den += __expf(t - mx);
          }
        }
        const float inv = 1.0f / den;
        const float mi  = mkf[i];
        const float rxi = rx[i], ryi = ry[i], rzi = rz[i];
        for (int c = 0; c < NNEI / 8; ++c) {
          v8h sv = srow[c];
          v8h av;
#pragma unroll
          for (int u = 0; u < 8; ++u) {
            const int j = c * 8 + u;
            float t   = ((float)sv[u] + ATTNW_SHIFT) * swi * swl[j] - ATTNW_SHIFT;
            float p   = __expf(t - mx) * inv;
            float ang = rxi * rx[j] + ryi * ry[j] + rzi * rz[j];
            av[u] = (_Float16)(p * mi * swi * swl[j] * ang);
          }
          arow[c] = av;
        }
        arow[15] = zz;                         // pad j = 120..127
      } else {
#pragma unroll
        for (int c = 0; c < 16; ++c) arow[c] = zz;
      }
    }
    __syncthreads();

    // prefetch residual rows for stage 7 (one byte per 128B line)
    for (int off = tid * 32; off < NNEI * EE; off += THREADS * 32)
      __builtin_prefetch(xrow + off, 0, 3);

    // ---- stage 6: o = attn @ v  (B^T = vT) -> oh in R2 (f16) ----
    {
      const int mt = wave;
      v16h a[4];
#pragma unroll
      for (int kk = 0; kk < 4; ++kk) a[kk] = frag_a(R1, mt * 16, kk * 32, NP, lane);
      for (int nt = 0; nt < HH / 16; ++nt) {
        v8f c = wmma_k128(a, R3, nt * 16, NP, lane);
        const int n  = nt * 16 + (lane & 15);
        const int mb = mt * 16 + ((lane >> 4) << 3);
#pragma unroll
        for (int vg = 0; vg < 8; ++vg) R2[(mb + vg) * EE + n] = (_Float16)c[vg];
      }
    }
    __syncthreads();
    weights_wait(wave);        // W_out^T in LDS now
    __syncthreads();

    // ---- stage 7: y = o @ W_out + b_out + residual -> ybuf (f32) ----
    {
      const int mt = wave;
      v16h a[4];
#pragma unroll
      for (int kk = 0; kk < 4; ++kk) a[kk] = frag_a(R2, mt * 16, kk * 32, EE, lane);
      for (int nt = 0; nt < EE / 16; ++nt) {
        v8f c = wmma_k128(a, WL, nt * 16, HH, lane);
        const int n  = nt * 16 + (lane & 15);
        const int mb = mt * 16 + ((lane >> 4) << 3);
        const float bn = bo[n];
#pragma unroll
        for (int vg = 0; vg < 8; ++vg) {
          const int m = mb + vg;
          const float res = (m < NNEI) ? xrow[m * EE + n] : 0.0f;
          ybuf[m * EE + n] = c[vg] + bn + res;
        }
      }
    }
    __syncthreads();

    // ---- stage 8: LayerNorm rows -> global output (16B vectors) ----
    if (tid < NNEI) {
      const v4f* yr4 = (const v4f*)(ybuf + tid * EE);
      float s1 = 0.f, s2 = 0.f;
#pragma unroll 8
      for (int c = 0; c < 32; ++c) {
        v4f v = yr4[c];
#pragma unroll
        for (int u = 0; u < 4; ++u) { s1 += v[u]; s2 = fmaf(v[u], v[u], s2); }
      }
      const float mu  = s1 * (1.0f / EE);
      const float var = s2 * (1.0f / EE) - mu * mu;
      const float inv = rsqrtf(var + LN_EPS);
      float* op = outp + (size_t)(b * NNEI + tid) * EE;
#pragma unroll 8
      for (int c = 0; c < 32; ++c) {
        v4f v = yr4[c];
        v4f g = *(const v4f*)(lng + c * 4);
        v4f bb = *(const v4f*)(lnb + c * 4);
        v4f o;
#pragma unroll
        for (int u = 0; u < 4; ++u) o[u] = (v[u] - mu) * inv * g[u] + bb[u];
        *(v4f*)(op + c * 4) = o;
      }
    }
  }
}

extern "C" void kernel_launch(void* const* d_in, const int* in_sizes, int n_in,
                              void* d_out, int out_size, void* d_ws, size_t ws_size,
                              hipStream_t stream) {
  const float* input_G  = (const float*)d_in[0];
  const int*   nei_mask = (const int*)d_in[1];
  const float* input_r  = (const float*)d_in[2];
  const float* swp      = (const float*)d_in[3];
  const float* W_in     = (const float*)d_in[4];
  const float* b_in     = (const float*)d_in[5];
  const float* W_out    = (const float*)d_in[6];
  const float* b_out    = (const float*)d_in[7];
  const float* ln_g     = (const float*)d_in[8];
  const float* ln_b     = (const float*)d_in[9];

  _Float16* wtin = (_Float16*)d_ws;                       // L*384*128 f16
  _Float16* wot  = wtin + (size_t)LAYERS * K3H * EE;      // L*128*128 f16

  const int prep_elems = LAYERS * K3H * EE + LAYERS * EE * HH; // 131072
  nga_prep<<<(prep_elems + 255) / 256, 256, 0, stream>>>(W_in, W_out, wtin, wot);

  nga_main<<<NB, THREADS, SMEM_BYTES, stream>>>(
      input_G, nei_mask, input_r, swp, b_in, b_out, ln_g, ln_b,
      wtin, wot, (float*)d_out);
}
